// PNAGnn_26027501814365
// MI455X (gfx1250) — compile-verified, matrix operand order checked by reading, not run on hardware
//
#include <hip/hip_runtime.h>
#include <hip/hip_bf16.h>
#include <math.h>

#define NN 50000
#define NB 500
#define NPG 100
#define NE 500000
#define HD 64

typedef __attribute__((ext_vector_type(16))) _Float16 v16h;
typedef __attribute__((ext_vector_type(8)))  float    v8f;

__device__ __forceinline__ float lrelu(float x){ return x > 0.f ? x : 0.01f * x; }
__device__ __forceinline__ float sigmoidf(float x){ return 1.f / (1.f + __expf(-x)); }

// Monotone-ordered float atomics (init mx=-inf, mn=+inf)
__device__ __forceinline__ void atomicMaxF(float* a, float v){
    if (v >= 0.f) atomicMax((int*)a, __float_as_int(v));
    else          atomicMin((unsigned int*)a, (unsigned int)__float_as_int(v));
}
__device__ __forceinline__ void atomicMinF(float* a, float v){
    if (v < 0.f)  atomicMax((unsigned int*)a, (unsigned int)__float_as_int(v));
    else          atomicMin((int*)a, __float_as_int(v));
}

// Build one 16x32 f16 fragment (A or B) from a row-major fp32 row.
// Element e<8 -> k = kbase + hi*8 + e ; e>=8 -> k = kbase + 16 + hi*8 + (e-8)
__device__ __forceinline__ v16h load_frag(const float* __restrict__ rowptr, int kbase, int hi){
    const float* p0 = rowptr + kbase + hi * 8;
    const float* p1 = p0 + 16;
    v16h r;
#pragma unroll
    for (int i = 0; i < 8; ++i){ r[i] = (_Float16)p0[i]; r[i + 8] = (_Float16)p1[i]; }
    return r;
}

// K=64 accumulation into acc (two chained WMMAs)
__device__ __forceinline__ v8f acc_k64(v8f acc, const float* __restrict__ arow,
                                       const float* __restrict__ wrow, int hi){
#pragma unroll
    for (int k0 = 0; k0 < 64; k0 += 32){
        v16h a = load_frag(arow, k0, hi);
        v16h b = load_frag(wrow, k0, hi);
        acc = __builtin_amdgcn_wmma_f32_16x16x32_f16(false, a, false, b, (short)0, acc, false, false);
    }
    return acc;
}

// ---------------- generic WMMA GEMM (K=64): C[M, ntiles*16] = A[M,64] @ W[Nout,64]^T + bias ----------
template<int ACT>
__global__ __launch_bounds__(256)
void k_wmma_gemm(const float* __restrict__ A, const float* __restrict__ W,
                 const float* __restrict__ bias, float* __restrict__ C,
                 int mtiles, int ntiles){
    int wid = (blockIdx.x * blockDim.x + threadIdx.x) >> 5;
    if (wid >= mtiles * ntiles) return;
    int lane = threadIdx.x & 31;
    int mt = wid / ntiles, nt = wid % ntiles;
    int m = lane & 15, hi = lane >> 4;
    const float* arow = A + (size_t)(mt * 16 + m) * 64;
    const float* wrow = W + (size_t)(nt * 16 + m) * 64;
    // two independent accumulators -> the two WMMAs pipeline (no D->C chain)
    v8f acc0 = {}, acc1 = {};
    {
        v16h a0 = load_frag(arow, 0, hi);
        v16h b0 = load_frag(wrow, 0, hi);
        v16h a1 = load_frag(arow, 32, hi);
        v16h b1 = load_frag(wrow, 32, hi);
        acc0 = __builtin_amdgcn_wmma_f32_16x16x32_f16(false, a0, false, b0, (short)0, acc0, false, false);
        acc1 = __builtin_amdgcn_wmma_f32_16x16x32_f16(false, a1, false, b1, (short)0, acc1, false, false);
    }
    int Nout = ntiles * 16;
    int col = nt * 16 + m;
    float bv = bias[col];
#pragma unroll
    for (int j = 0; j < 8; ++j){
        int row = mt * 16 + hi * 8 + j;
        float v = acc0[j] + acc1[j] + bv;
        if (ACT == 1) v = lrelu(v);
        else if (ACT == 2) v = fmaxf(v, 0.f);
        C[(size_t)row * Nout + col] = v;
    }
}

// ------------- PNA edge message (layers 2,3): msg = cat[h[dst],h[src]] @ pw^T + pb, fused scatter ----
__global__ __launch_bounds__(256)
void k_msg_wmma(const float* __restrict__ hin, const int* __restrict__ ei,
                const float* __restrict__ pw, const float* __restrict__ pb,
                float* __restrict__ s, float* __restrict__ sq,
                float* __restrict__ mx, float* __restrict__ mn){
    const int etiles = NE / 16;
    int wid = (blockIdx.x * blockDim.x + threadIdx.x) >> 5;
    if (wid >= etiles * 4) return;
    int lane = threadIdx.x & 31;
    int et = wid >> 2, nt = wid & 3;
    int m = lane & 15, hi = lane >> 4;
    int e_m  = et * 16 + m;
    int dstm = ei[NE + e_m];
    int srcm = ei[e_m];
    const float* wrow = pw + (size_t)(nt * 16 + m) * 128;
    const float* drow = hin + (size_t)dstm * HD;
    const float* srow = hin + (size_t)srcm * HD;
    // two independent accumulator chains (dst half / src half of K=128)
    v8f acc0 = {}, acc1 = {};
#pragma unroll
    for (int t = 0; t < 2; ++t){
        v16h ad = load_frag(drow, t * 32, hi);
        v16h bd = load_frag(wrow, t * 32, hi);
        acc0 = __builtin_amdgcn_wmma_f32_16x16x32_f16(false, ad, false, bd, (short)0, acc0, false, false);
        v16h as = load_frag(srow, t * 32, hi);
        v16h bs = load_frag(wrow, 64 + t * 32, hi);
        acc1 = __builtin_amdgcn_wmma_f32_16x16x32_f16(false, as, false, bs, (short)0, acc1, false, false);
    }
    int col = nt * 16 + m;
    float bv = pb[col];
#pragma unroll
    for (int j = 0; j < 8; ++j){
        int e = et * 16 + hi * 8 + j;
        int d = ei[NE + e];
        float v = acc0[j] + acc1[j] + bv;
        size_t idx = (size_t)d * HD + col;
        atomicAdd(&s[idx], v);
        atomicAdd(&sq[idx], v * v);
        atomicMaxF(&mx[idx], v);
        atomicMinF(&mn[idx], v);
    }
}

// ------------- PNA post_nn (F=64): out = [h, agg, s1*agg, s2*agg] @ qw^T + qb -------------
__global__ __launch_bounds__(256)
void k_post_wmma(const float* __restrict__ hin,
                 const float* __restrict__ b0, const float* __restrict__ b1,
                 const float* __restrict__ b2, const float* __restrict__ b3,
                 const float* __restrict__ qw, const float* __restrict__ qb,
                 const float* __restrict__ s1, const float* __restrict__ s2,
                 float* __restrict__ out){
    const int mtiles = NN / 16;
    int wid = (blockIdx.x * blockDim.x + threadIdx.x) >> 5;
    if (wid >= mtiles * 4) return;
    int lane = threadIdx.x & 31;
    int mt = wid >> 2, nt = wid & 3;
    int m = lane & 15, hi = lane >> 4;
    int arow0 = mt * 16 + m;
    const float* wrow = qw + (size_t)(nt * 16 + m) * 832;
    const float* aggp[4] = { b0, b1, b2, b3 };
    // four independent accumulator chains (h / agg / s1*agg / s2*agg column blocks)
    v8f am = {}, aa = {}, ab = {}, ac = {};
    am = acc_k64(am, hin + (size_t)arow0 * HD, wrow, hi);
#pragma unroll
    for (int g = 0; g < 4; ++g){
        const float* arow = aggp[g] + (size_t)arow0 * HD;
        aa = acc_k64(aa, arow, wrow + 64  + g * 64, hi);
        ab = acc_k64(ab, arow, wrow + 320 + g * 64, hi);
        ac = acc_k64(ac, arow, wrow + 576 + g * 64, hi);
    }
    int col = nt * 16 + m;
    float bv = qb[col];
#pragma unroll
    for (int j = 0; j < 8; ++j){
        int row = mt * 16 + hi * 8 + j;
        float v = am[j] + aa[j] + s1[row] * ab[j] + s2[row] * ac[j] + bv;
        out[(size_t)row * HD + col] = v;
    }
}

// ---------------- small helper kernels ----------------
__global__ void k_fill(float* p, float v, int n){
    int i = blockIdx.x * blockDim.x + threadIdx.x;
    if (i < n) p[i] = v;
}
__global__ void k_count(const int* __restrict__ ei, float* __restrict__ counts){
    int e = blockIdx.x * blockDim.x + threadIdx.x;
    if (e >= NE) return;
    atomicAdd(&counts[ei[NE + e]], 1.f);
}
__global__ void k_deg(const float* __restrict__ counts, float* __restrict__ degc, float* avgacc){
    __shared__ float sh[256];
    int n = blockIdx.x * 256 + threadIdx.x;
    float v = 0.f;
    if (n < NN){
        float c = counts[n];
        degc[n] = fmaxf(c, 1.f);
        v = __logf(c + 1.f);
    }
    sh[threadIdx.x] = v; __syncthreads();
    for (int o = 128; o > 0; o >>= 1){
        if (threadIdx.x < o) sh[threadIdx.x] += sh[threadIdx.x + o];
        __syncthreads();
    }
    if (threadIdx.x == 0) atomicAdd(avgacc, sh[0]);
}
__global__ void k_scale(const float* __restrict__ degc, const float* __restrict__ avgacc,
                        float* __restrict__ s1, float* __restrict__ s2){
    int n = blockIdx.x * blockDim.x + threadIdx.x;
    if (n >= NN) return;
    float avg = avgacc[0] / (float)NN;
    float ld = __logf(degc[n] + 1.f);
    s1[n] = ld / avg;
    s2[n] = avg / ld;
}
// layer-1 edge messages (F_in = 2), scalar
__global__ void k_msg1(const float* __restrict__ x, const int* __restrict__ ei,
                       const float* __restrict__ pw, const float* __restrict__ pb,
                       float* __restrict__ s, float* __restrict__ sq,
                       float* __restrict__ mx, float* __restrict__ mn){
    int e = blockIdx.x * blockDim.x + threadIdx.x;
    if (e >= NE) return;
    int srci = ei[e], dsti = ei[NE + e];
    float xi0 = x[dsti * 2], xi1 = x[dsti * 2 + 1];
    float xj0 = x[srci * 2], xj1 = x[srci * 2 + 1];
#pragma unroll
    for (int f = 0; f < 2; ++f){
        float v = pw[f*4]*xi0 + pw[f*4+1]*xi1 + pw[f*4+2]*xj0 + pw[f*4+3]*xj1 + pb[f];
        int idx = dsti * 2 + f;
        atomicAdd(&s[idx], v);
        atomicAdd(&sq[idx], v * v);
        atomicMaxF(&mx[idx], v);
        atomicMinF(&mn[idx], v);
    }
}
__global__ void k_finalize(float* s, float* sq, float* mx, float* mn,
                           const float* __restrict__ counts, const float* __restrict__ degc, int F){
    int i = blockIdx.x * blockDim.x + threadIdx.x;
    if (i >= NN * F) return;
    int n = i / F;
    float dc = degc[n];
    float mean = s[i] / dc;
    float var = sq[i] / dc - mean * mean;
    if (var < 0.f) var = 0.f;
    s[i]  = mean;
    sq[i] = sqrtf(var + 1e-5f);
    bool has = counts[n] > 0.f;
    mx[i] = has ? mx[i] : 0.f;
    mn[i] = has ? mn[i] : 0.f;
}
// layer-1 post_nn (input dim 26), scalar
__global__ void k_post1(const float* __restrict__ x,
                        const float* __restrict__ b0, const float* __restrict__ b1,
                        const float* __restrict__ b2, const float* __restrict__ b3,
                        const float* __restrict__ qw, const float* __restrict__ qb,
                        const float* __restrict__ s1v, const float* __restrict__ s2v,
                        float* __restrict__ out){
    int i = blockIdx.x * blockDim.x + threadIdx.x;
    if (i >= NN * HD) return;
    int n = i >> 6, j = i & 63;
    const float* w = qw + j * 26;
    float sa = s1v[n], sb = s2v[n];
    float acc = qb[j] + w[0] * x[n*2] + w[1] * x[n*2+1];
    const float* bufs[4] = { b0, b1, b2, b3 };
#pragma unroll
    for (int g = 0; g < 4; ++g){
#pragma unroll
        for (int f = 0; f < 2; ++f){
            float a = bufs[g][n * 2 + f];
            int c = 2 + g * 2 + f;
            acc += w[c] * a + w[c + 8] * (a * sa) + w[c + 16] * (a * sb);
        }
    }
    out[i] = acc;
}
// GRU1 input gates (input dim 2), scalar
__global__ void k_gi1(const float* __restrict__ x, const float* __restrict__ wih,
                      const float* __restrict__ bih, float* __restrict__ gi){
    int i = blockIdx.x * blockDim.x + threadIdx.x;
    if (i >= NN * 192) return;
    int n = i / 192, j = i % 192;
    gi[i] = wih[j*2] * x[n*2] + wih[j*2+1] * x[n*2+1] + bih[j];
}
__global__ void k_gru(const float* __restrict__ gi, const float* __restrict__ gh,
                      const float* __restrict__ h, float* __restrict__ state){
    int i = blockIdx.x * blockDim.x + threadIdx.x;
    if (i >= NN * HD) return;
    int n = i >> 6, k = i & 63;
    size_t b = (size_t)n * 192 + k;
    float r  = sigmoidf(gi[b]       + gh[b]);
    float z  = sigmoidf(gi[b + 64]  + gh[b + 64]);
    float nn = tanhf  (gi[b + 128] + r * gh[b + 128]);
    state[i] = (1.f - z) * nn + z * h[i];
}
// node readout layers 2,3
__global__ void k_t2(const float* __restrict__ t1, const float* __restrict__ w,
                     const float* __restrict__ b, float* __restrict__ t2){
    int i = blockIdx.x * blockDim.x + threadIdx.x;
    if (i >= NN * 8) return;
    int n = i >> 3, j = i & 7;
    const float* wr = w + j * 64;
    const float* xr = t1 + (size_t)n * 64;
    float a = b[j];
    for (int k = 0; k < 64; ++k) a += wr[k] * xr[k];
    t2[i] = lrelu(a);
}
__global__ void k_ynode(const float* __restrict__ t2, const float* __restrict__ w,
                        const float* __restrict__ b, float* __restrict__ out){
    int i = blockIdx.x * blockDim.x + threadIdx.x;
    if (i >= NN * 3) return;
    int n = i / 3, j = i % 3;
    const float* wr = w + j * 8;
    const float* xr = t2 + n * 8;
    float a = b[j];
    for (int k = 0; k < 8; ++k) a += wr[k] * xr[k];
    out[i] = lrelu(a);
}
// fused Set2Set step: one block per graph
__global__ void k_s2s_step(const float* __restrict__ xs,
                           const float* __restrict__ lwih, const float* __restrict__ lbih,
                           const float* __restrict__ lwhh, const float* __restrict__ lbhh,
                           float* __restrict__ hbuf, float* __restrict__ cbuf,
                           float* __restrict__ qstar){
    __shared__ float sg[256];
    __shared__ float sq[64];
    __shared__ float se[NPG];
    __shared__ float red[2];
    int b = blockIdx.x, tid = threadIdx.x;
    {   // LSTM gates: 256 outputs, K = 128 + 64
        float a = lbih[tid] + lbhh[tid];
        const float* wi = lwih + tid * 128;
        const float* qs = qstar + b * 128;
        for (int k = 0; k < 128; ++k) a += wi[k] * qs[k];
        const float* wh = lwhh + tid * 64;
        const float* hb = hbuf + b * 64;
        for (int k = 0; k < 64; ++k) a += wh[k] * hb[k];
        sg[tid] = a;
    }
    __syncthreads();
    if (tid < 64){
        float ii = sg[tid], ff = sg[64 + tid], gg = sg[128 + tid], oo = sg[192 + tid];
        float c2 = sigmoidf(ff) * cbuf[b * 64 + tid] + sigmoidf(ii) * tanhf(gg);
        float q = sigmoidf(oo) * tanhf(c2);
        cbuf[b * 64 + tid] = c2;
        hbuf[b * 64 + tid] = q;
        sq[tid] = q;
    }
    __syncthreads();
    if (tid < NPG){
        const float* xr = xs + (size_t)(b * NPG + tid) * HD;
        float ev = 0.f;
        for (int k = 0; k < 64; ++k) ev += xr[k] * sq[k];
        se[tid] = ev;
    }
    __syncthreads();
    if (tid == 0){ float m = -1e30f; for (int n = 0; n < NPG; ++n) m = fmaxf(m, se[n]); red[0] = m; }
    __syncthreads();
    if (tid < NPG) se[tid] = __expf(se[tid] - red[0]);
    __syncthreads();
    if (tid == 0){ float s = 0.f; for (int n = 0; n < NPG; ++n) s += se[n]; red[1] = s; }
    __syncthreads();
    if (tid < 64){
        float inv = 1.f / red[1];
        float r = 0.f;
        for (int n = 0; n < NPG; ++n) r += se[n] * xs[(size_t)(b * NPG + n) * HD + tid];
        qstar[b * 128 + tid] = sq[tid];
        qstar[b * 128 + 64 + tid] = r * inv;
    }
}
__global__ void k_graph_readout(const float* __restrict__ qstar,
                                const float* __restrict__ w1, const float* __restrict__ b1,
                                const float* __restrict__ w2, const float* __restrict__ b2,
                                const float* __restrict__ w3, const float* __restrict__ b3,
                                float* __restrict__ out){
    __shared__ float h1[64];
    __shared__ float h2[64];
    int b = blockIdx.x, tid = threadIdx.x;
    {
        float a = b1[tid];
        const float* q = qstar + b * 128;
        const float* wr = w1 + tid * 128;
        for (int k = 0; k < 128; ++k) a += wr[k] * q[k];
        h1[tid] = fmaxf(a, 0.f);
    }
    __syncthreads();
    {
        float a = b2[tid];
        const float* wr = w2 + tid * 64;
        for (int k = 0; k < 64; ++k) a += wr[k] * h1[k];
        h2[tid] = fmaxf(a, 0.f);
    }
    __syncthreads();
    if (tid < 3){
        float a = b3[tid];
        const float* wr = w3 + tid * 64;
        for (int k = 0; k < 64; ++k) a += wr[k] * h2[k];
        out[NN * 3 + b * 3 + tid] = lrelu(a);
    }
}
__global__ void k_sqdiff(const float* __restrict__ a, const float* __restrict__ t,
                         int n, float* acc){
    __shared__ float sh[256];
    int i = blockIdx.x * 256 + threadIdx.x;
    float v = 0.f;
    if (i < n){ float d = a[i] - t[i]; v = d * d; }
    sh[threadIdx.x] = v; __syncthreads();
    for (int o = 128; o > 0; o >>= 1){
        if (threadIdx.x < o) sh[threadIdx.x] += sh[threadIdx.x + o];
        __syncthreads();
    }
    if (threadIdx.x == 0) atomicAdd(acc, sh[0]);
}
__global__ void k_loss_final(const float* __restrict__ lacc, float* __restrict__ out){
    out[0] = 0.5f * (lacc[0] / (float)(NN * 3) + lacc[1] / (float)(NB * 3));
}

static inline int cdiv(int a, int b){ return (a + b - 1) / b; }

extern "C" void kernel_launch(void* const* d_in, const int* in_sizes, int n_in,
                              void* d_out, int out_size, void* d_ws, size_t ws_size,
                              hipStream_t stream){
    const float* x  = (const float*)d_in[0];
    const int*   ei = (const int*)d_in[1];
    const float* tn = (const float*)d_in[2];
    const float* tg = (const float*)d_in[3];
    // d_in[4] = batch (unused: graphs are contiguous blocks of NPG nodes)
    const float* c1_pre_w  = (const float*)d_in[5];
    const float* c1_pre_b  = (const float*)d_in[6];
    const float* c1_post_w = (const float*)d_in[7];
    const float* c1_post_b = (const float*)d_in[8];
    const float* c1_lin_w  = (const float*)d_in[9];
    const float* c1_lin_b  = (const float*)d_in[10];
    const float* c_pre_w   = (const float*)d_in[11];
    const float* c_pre_b   = (const float*)d_in[12];
    const float* c_post_w  = (const float*)d_in[13];
    const float* c_post_b  = (const float*)d_in[14];
    const float* c_lin_w   = (const float*)d_in[15];
    const float* c_lin_b   = (const float*)d_in[16];
    const float* g1_wih = (const float*)d_in[17];
    const float* g1_bih = (const float*)d_in[18];
    const float* g1_whh = (const float*)d_in[19];
    const float* g1_bhh = (const float*)d_in[20];
    const float* g_wih  = (const float*)d_in[21];
    const float* g_bih  = (const float*)d_in[22];
    const float* g_whh  = (const float*)d_in[23];
    const float* g_bhh  = (const float*)d_in[24];
    const float* l_wih  = (const float*)d_in[25];
    const float* l_bih  = (const float*)d_in[26];
    const float* l_whh  = (const float*)d_in[27];
    const float* l_bhh  = (const float*)d_in[28];
    const float* n_w1 = (const float*)d_in[29];
    const float* n_b1 = (const float*)d_in[30];
    const float* n_w2 = (const float*)d_in[31];
    const float* n_b2 = (const float*)d_in[32];
    const float* n_w3 = (const float*)d_in[33];
    const float* n_b3 = (const float*)d_in[34];
    const float* gr_w1 = (const float*)d_in[35];
    const float* gr_b1 = (const float*)d_in[36];
    const float* gr_w2 = (const float*)d_in[37];
    const float* gr_b2 = (const float*)d_in[38];
    const float* gr_w3 = (const float*)d_in[39];
    const float* gr_b3 = (const float*)d_in[40];

    float* out = (float*)d_out;
    float* ws  = (float*)d_ws;

    const size_t NH = (size_t)NN * HD;
    float* state = ws;
    float* b0 = ws + 1 * NH;          // sum -> mean
    float* b1 = ws + 2 * NH;          // sumsq -> std
    float* b2 = ws + 3 * NH;          // max
    float* b3 = ws + 4 * NH;          // min
    float* post = ws + 5 * NH;
    float* ybuf = ws + 6 * NH;
    float* gi   = ws + 7 * NH;        // N x 192
    float* gh   = ws + 10 * NH;       // N x 192
    float* counts = ws + 13 * NH;
    float* degc = counts + NN;
    float* s1 = degc + NN;
    float* s2 = s1 + NN;
    float* avgacc = s2 + NN;          // [0]=avg log acc, [1],[2]=loss acc
    float* lacc = avgacc + 1;
    float* hbuf = avgacc + 4;
    float* cbuf = hbuf + NB * 64;
    float* qstar = cbuf + NB * 64;

    const float INF = __builtin_inff();
    const int T = 256;

    auto gemm0 = [&](const float* A, const float* W, const float* bias, float* C,
                     int mtiles, int ntiles){
        k_wmma_gemm<0><<<cdiv(mtiles * ntiles * 32, T), T, 0, stream>>>(A, W, bias, C, mtiles, ntiles);
    };

    // degree stats
    k_fill<<<cdiv(NN, T), T, 0, stream>>>(counts, 0.f, NN);
    k_fill<<<1, 32, 0, stream>>>(avgacc, 0.f, 3);
    k_count<<<cdiv(NE, T), T, 0, stream>>>(ei, counts);
    k_deg<<<cdiv(NN, T), T, 0, stream>>>(counts, degc, avgacc);
    k_scale<<<cdiv(NN, T), T, 0, stream>>>(degc, avgacc, s1, s2);

    // ---- PNA layer 1 (F_in = 2) ----
    k_fill<<<cdiv(NN*2, T), T, 0, stream>>>(b0, 0.f, NN*2);
    k_fill<<<cdiv(NN*2, T), T, 0, stream>>>(b1, 0.f, NN*2);
    k_fill<<<cdiv(NN*2, T), T, 0, stream>>>(b2, -INF, NN*2);
    k_fill<<<cdiv(NN*2, T), T, 0, stream>>>(b3,  INF, NN*2);
    k_msg1<<<cdiv(NE, T), T, 0, stream>>>(x, ei, c1_pre_w, c1_pre_b, b0, b1, b2, b3);
    k_finalize<<<cdiv(NN*2, T), T, 0, stream>>>(b0, b1, b2, b3, counts, degc, 2);
    k_post1<<<cdiv(NN*HD, T), T, 0, stream>>>(x, b0, b1, b2, b3, c1_post_w, c1_post_b, s1, s2, post);
    gemm0(post, c1_lin_w, c1_lin_b, ybuf, NN/16, 4);
    k_gi1<<<cdiv(NN*192, T), T, 0, stream>>>(x, g1_wih, g1_bih, gi);
    gemm0(ybuf, g1_whh, g1_bhh, gh, NN/16, 12);
    k_gru<<<cdiv(NN*HD, T), T, 0, stream>>>(gi, gh, ybuf, state);

    // ---- PNA layers 2..3 (F_in = 64) ----
    for (int L = 0; L < 2; ++L){
        k_fill<<<cdiv((int)NH, T), T, 0, stream>>>(b0, 0.f, (int)NH);
        k_fill<<<cdiv((int)NH, T), T, 0, stream>>>(b1, 0.f, (int)NH);
        k_fill<<<cdiv((int)NH, T), T, 0, stream>>>(b2, -INF, (int)NH);
        k_fill<<<cdiv((int)NH, T), T, 0, stream>>>(b3,  INF, (int)NH);
        k_msg_wmma<<<cdiv((NE/16) * 4 * 32, T), T, 0, stream>>>(state, ei, c_pre_w, c_pre_b, b0, b1, b2, b3);
        k_finalize<<<cdiv((int)NH, T), T, 0, stream>>>(b0, b1, b2, b3, counts, degc, HD);
        k_post_wmma<<<cdiv((NN/16) * 4 * 32, T), T, 0, stream>>>(state, b0, b1, b2, b3,
                                                                 c_post_w, c_post_b, s1, s2, post);
        gemm0(post, c_lin_w, c_lin_b, ybuf, NN/16, 4);
        gemm0(state, g_wih, g_bih, gi, NN/16, 12);
        gemm0(ybuf,  g_whh, g_bhh, gh, NN/16, 12);
        k_gru<<<cdiv(NN*HD, T), T, 0, stream>>>(gi, gh, ybuf, state);
    }

    // ---- node readout ----
    k_wmma_gemm<1><<<cdiv((NN/16) * 4 * 32, T), T, 0, stream>>>(state, n_w1, n_b1, b0, NN/16, 4);
    k_t2<<<cdiv(NN*8, T), T, 0, stream>>>(b0, n_w2, n_b2, b1);
    k_ynode<<<cdiv(NN*3, T), T, 0, stream>>>(b1, n_w3, n_b3, out);

    // ---- Set2Set (100 sequential fused steps) ----
    k_fill<<<cdiv(NB*64, T), T, 0, stream>>>(hbuf, 0.f, NB*64);
    k_fill<<<cdiv(NB*64, T), T, 0, stream>>>(cbuf, 0.f, NB*64);
    k_fill<<<cdiv(NB*128, T), T, 0, stream>>>(qstar, 0.f, NB*128);
    for (int step = 0; step < NPG; ++step)
        k_s2s_step<<<NB, 256, 0, stream>>>(state, l_wih, l_bih, l_whh, l_bhh, hbuf, cbuf, qstar);

    // ---- graph readout ----
    k_graph_readout<<<NB, 64, 0, stream>>>(qstar, gr_w1, gr_b1, gr_w2, gr_b2, gr_w3, gr_b3, out);

    // ---- losses ----
    k_sqdiff<<<cdiv(NN*3, T), T, 0, stream>>>(out, tn, NN*3, lacc);
    k_sqdiff<<<cdiv(NB*3, T), T, 0, stream>>>(out + NN*3, tg, NB*3, lacc + 1);
    k_loss_final<<<1, 1, 0, stream>>>(lacc, out + NN*3 + NB*3);
}
